// MonotonicNormalizer_90967407329416
// MI455X (gfx1250) — matrix operand order repaired and checked by reading, not compiled, not run on hardware
//
#include <hip/hip_runtime.h>
#include <hip/hip_bf16.h>

// ---------------- problem constants ----------------
#define BATCH   512
#define DDIM    32
#define CONDN   30
#define SSTEPS  21                      // NB_STEPS + 1
#define BD      (BATCH * DDIM)          // 16384
#define NGRID   (BATCH * SSTEPS * DDIM) // 344064 quadrature points
#define NPTS    (NGRID + BD)            // 360448 total MLP evaluations

// tile config for the main kernel
#define ROWS_PER_BLOCK 48               // 3 m-tiles of 16
#define ACT_STRIDE     264              // 256 + 8 bf16 pad (528B: 16B aligned, bank-conflict free)
#define IN0_STRIDE     40               // 32 + 8 bf16 pad (80B aligned)

typedef __attribute__((ext_vector_type(16))) __bf16 v16bf;
typedef __attribute__((ext_vector_type(8)))  float  v8f;

union FragU { uint4 u[2]; v16bf v; };

__device__ __forceinline__ float bf2f(unsigned short u) {
    union { unsigned v; float f; } c; c.v = ((unsigned)u) << 16; return c.f;
}

// A-fragment (16x32 bf16, M x K) from LDS, row-major activation buffer.
// lanes 0-15: M=lane,   K = {0..7, 16..23}  (two b128 loads)
// lanes 16-31: M=lane-16, K = {8..15, 24..31}
__device__ __forceinline__ v16bf ld_afrag(const __bf16* buf, int stride,
                                          int mBase, int kBase, int lane) {
    const __bf16* p = buf + (mBase + (lane & 15)) * stride + kBase + ((lane >> 4) << 3);
    FragU f;
    f.u[0] = *(const uint4*)(p);
    f.u[1] = *(const uint4*)(p + 16);
    return f.v;
}

// B-fragment (32x16 bf16, K x N) from pre-swizzled global fragment buffer.
// Each fragment = 32 lanes * 16 bf16 (1KB), lane-major; 2 coalesced b128 loads.
__device__ __forceinline__ v16bf ld_bfrag(const __bf16* wf, int fragIdx, int lane) {
    const uint4* p = (const uint4*)(wf + fragIdx * 512) + lane * 2;
    FragU f;
    f.u[0] = p[0];
    f.u[1] = p[1];
    return f.v;
}

// ---------------- prep: weight bf16 swizzle + CC weights ----------------
__global__ void umnn_prep(const float* __restrict__ W0, const float* __restrict__ W1,
                          const float* __restrict__ W2,
                          __bf16* __restrict__ W0f, __bf16* __restrict__ W1f,
                          __bf16* __restrict__ W2f, float* __restrict__ ccw) {
    int idx = blockIdx.x * 256 + threadIdx.x;   // 65536 threads
    if (idx < 256 * 256) {
        int k = idx >> 8, n = idx & 255;
        int kt = k >> 5, kin = k & 31, nt = n >> 4;
        int lane = (((kin >> 3) & 1) << 4) | (n & 15);
        int e = (kin & 7) | ((kin >> 4) << 3);
        int dst = (nt * 8 + kt) * 512 + lane * 16 + e;
        W1f[dst] = (__bf16)W1[idx];
        W2f[dst] = (__bf16)W2[idx];
    }
    if (idx < 32 * 256) {                       // W0 padded K 31 -> 32
        int k = idx >> 8, n = idx & 255;
        float v = (k < 31) ? W0[k * 256 + n] : 0.f;
        int nt = n >> 4;
        int lane = (((k >> 3) & 1) << 4) | (n & 15);
        int e = (k & 7) | ((k >> 4) << 3);
        W0f[nt * 512 + lane * 16 + e] = (__bf16)v;
    }
    if (idx < SSTEPS) {                         // Clenshaw-Curtis weights
        const double n = 20.0;
        double colscale = (idx == 0 || idx == 20) ? 0.5 : 1.0;
        double s = 0.0;
        for (int i = 0; i <= 20; i += 2) {
            double Wi = (i == 0) ? 1.0 : 2.0 / (1.0 - (double)i * (double)i);
            s += cos((double)i * (double)idx * 3.14159265358979323846 / n) * Wi;
        }
        ccw[idx] = (float)(s * (2.0 / n) * colscale);
    }
}

// ---------------- dense 256->256 layer (WMMA bf16) ----------------
__device__ __forceinline__ void dense_layer(const __bf16* __restrict__ bufIn,
                                            __bf16* __restrict__ bufOut,
                                            const __bf16* __restrict__ Wf,
                                            const float* __restrict__ bias,
                                            int wave, int lane) {
#pragma unroll
    for (int nti = 0; nti < 4; ++nti) {
        const int nt = wave * 4 + nti;
        const int n  = nt * 16 + (lane & 15);
        const float bv = bias[n];
        v8f acc[3];
#pragma unroll
        for (int mt = 0; mt < 3; ++mt)
            acc[mt] = (v8f){bv, bv, bv, bv, bv, bv, bv, bv};
#pragma unroll
        for (int kt = 0; kt < 8; ++kt) {
            v16bf bf = ld_bfrag(Wf, nt * 8 + kt, lane);
#pragma unroll
            for (int mt = 0; mt < 3; ++mt) {
                v16bf af = ld_afrag(bufIn, ACT_STRIDE, mt * 16, kt * 32, lane);
                acc[mt] = __builtin_amdgcn_wmma_f32_16x16x32_bf16(
                    false, af, false, bf, (short)0, acc[mt], false, false);
            }
        }
        // fused ReLU + bf16 pack + store (C layout: lane -> fixed N, VGPR v -> M)
        const int mrow0 = (lane >> 4) << 3;
#pragma unroll
        for (int mt = 0; mt < 3; ++mt)
#pragma unroll
            for (int v = 0; v < 8; ++v) {
                float r = fmaxf(acc[mt][v], 0.f);
                bufOut[(mt * 16 + mrow0 + v) * ACT_STRIDE + n] = (__bf16)r;
            }
    }
}

// ---------------- main MLP kernel ----------------
__global__ __launch_bounds__(128)
void umnn_mlp(const float* __restrict__ x, const float* __restrict__ h,
              const float* __restrict__ b0, const float* __restrict__ b1,
              const float* __restrict__ b2, const float* __restrict__ W3,
              const float* __restrict__ b3,
              const __bf16* __restrict__ W0f, const __bf16* __restrict__ W1f,
              const __bf16* __restrict__ W2f,
              float* __restrict__ F, float* __restrict__ outJac) {
    __shared__ __align__(16) __bf16 in0[ROWS_PER_BLOCK * IN0_STRIDE];
    __shared__ __align__(16) __bf16 bufA[ROWS_PER_BLOCK * ACT_STRIDE];
    __shared__ __align__(16) __bf16 bufB[ROWS_PER_BLOCK * ACT_STRIDE];
    __shared__ __align__(16) float  w3s[256];

    const int t    = threadIdx.x;      // 128 threads = 4 waves (wave32)
    const int lane = t & 31;
    const int wave = t >> 5;
    const int blockBase = blockIdx.x * ROWS_PER_BLOCK;

    // stage W3 (f32) into LDS
    w3s[t]       = W3[t];
    w3s[t + 128] = W3[t + 128];

    // ---- build the 48 x 32 bf16 input tile: [X | h(b,d,0..29) | 0] ----
    for (int e = t; e < ROWS_PER_BLOCK * 32; e += 128) {
        const int row = e >> 5;
        const int col = e & 31;
        const int p = blockBase + row;
        float v = 0.f;
        if (p < NPTS) {
            int bd; float scale;
            if (p < NGRID) {
                const int d  = p & 31;
                const int bs = p >> 5;
                const int s  = bs % SSTEPS;
                const int b  = bs / SSTEPS;
                bd = b * DDIM + d;
                scale = (cosf((float)s * 3.14159265358979323846f / 20.f) + 1.f) * 0.5f;
            } else {
                bd = p - NGRID;
                scale = 1.f;
            }
            if (col == 0)       v = x[bd] * scale;
            else if (col <= 30) v = h[bd * CONDN + (col - 1)];
        }
        in0[row * IN0_STRIDE + col] = (__bf16)v;
    }
    __syncthreads();

    // ---- layer 0: 31(->32) -> 256, ReLU ----
#pragma unroll
    for (int nti = 0; nti < 4; ++nti) {
        const int nt = wave * 4 + nti;
        const int n  = nt * 16 + (lane & 15);
        const float bv = b0[n];
        v8f acc[3];
#pragma unroll
        for (int mt = 0; mt < 3; ++mt)
            acc[mt] = (v8f){bv, bv, bv, bv, bv, bv, bv, bv};
        v16bf bf = ld_bfrag(W0f, nt, lane);
#pragma unroll
        for (int mt = 0; mt < 3; ++mt) {
            v16bf af = ld_afrag(in0, IN0_STRIDE, mt * 16, 0, lane);
            acc[mt] = __builtin_amdgcn_wmma_f32_16x16x32_bf16(
                false, af, false, bf, (short)0, acc[mt], false, false);
        }
        const int mrow0 = (lane >> 4) << 3;
#pragma unroll
        for (int mt = 0; mt < 3; ++mt)
#pragma unroll
            for (int v = 0; v < 8; ++v) {
                float r = fmaxf(acc[mt][v], 0.f);
                bufA[(mt * 16 + mrow0 + v) * ACT_STRIDE + n] = (__bf16)r;
            }
    }
    __syncthreads();

    // ---- layers 1 & 2: 256 -> 256, ReLU ----
    dense_layer(bufA, bufB, W1f, b1, wave, lane);
    __syncthreads();
    dense_layer(bufB, bufA, W2f, b2, wave, lane);
    __syncthreads();

    // ---- layer 3: 256 -> 1, ELU(x)+1; VALU dot (0.1% of FLOPs) ----
    if (t < ROWS_PER_BLOCK) {
        const int p = blockBase + t;
        float acc = b3[0];
        const uint4* rp = (const uint4*)(bufA + t * ACT_STRIDE);
#pragma unroll 4
        for (int i = 0; i < 32; ++i) {
            uint4 q = rp[i];
            const float* w = w3s + i * 8;
            acc += bf2f((unsigned short)(q.x & 0xffff)) * w[0];
            acc += bf2f((unsigned short)(q.x >> 16))    * w[1];
            acc += bf2f((unsigned short)(q.y & 0xffff)) * w[2];
            acc += bf2f((unsigned short)(q.y >> 16))    * w[3];
            acc += bf2f((unsigned short)(q.z & 0xffff)) * w[4];
            acc += bf2f((unsigned short)(q.z >> 16))    * w[5];
            acc += bf2f((unsigned short)(q.w & 0xffff)) * w[6];
            acc += bf2f((unsigned short)(q.w >> 16))    * w[7];
        }
        const float f = (acc > 0.f) ? (acc + 1.f) : expf(acc);  // elu(x)+1
        if (p < NPTS) {
            if (p < NGRID) F[p] = f;
            else           outJac[p - NGRID] = f;
        }
    }
}

// ---------------- CC quadrature reduction -> z ----------------
__global__ void umnn_reduce(const float* __restrict__ x, const float* __restrict__ h,
                            const float* __restrict__ F, const float* __restrict__ ccw,
                            float* __restrict__ outZ) {
    const int bd = blockIdx.x * 256 + threadIdx.x;
    if (bd >= BD) return;
    const int b = bd >> 5, d = bd & 31;
    float s = 0.f;
#pragma unroll
    for (int si = 0; si < SSTEPS; ++si)
        s += ccw[si] * F[(b * SSTEPS + si) * DDIM + d];
    outZ[bd] = s * x[bd] * 0.5f + h[bd * CONDN];   // + h[b,d,0]
}

// ---------------- launcher ----------------
extern "C" void kernel_launch(void* const* d_in, const int* in_sizes, int n_in,
                              void* d_out, int out_size, void* d_ws, size_t ws_size,
                              hipStream_t stream) {
    const float* x  = (const float*)d_in[0];
    const float* h  = (const float*)d_in[1];
    const float* W0 = (const float*)d_in[2];
    const float* b0 = (const float*)d_in[3];
    const float* W1 = (const float*)d_in[4];
    const float* b1 = (const float*)d_in[5];
    const float* W2 = (const float*)d_in[6];
    const float* b2 = (const float*)d_in[7];
    const float* W3 = (const float*)d_in[8];
    const float* b3 = (const float*)d_in[9];

    float* outZ = (float*)d_out;
    float* outJ = outZ + BD;

    // workspace layout
    char* ws = (char*)d_ws;
    __bf16* W0f = (__bf16*)(ws);                          //  16 KB
    __bf16* W1f = (__bf16*)(ws + 16384);                  // 128 KB
    __bf16* W2f = (__bf16*)(ws + 16384 + 131072);         // 128 KB
    float*  ccw = (float*)(ws + 16384 + 2 * 131072);      //  <128 B
    float*  F   = (float*)(ws + 16384 + 2 * 131072 + 128);// 1.31 MB integrand values

    umnn_prep<<<256, 256, 0, stream>>>(W0, W1, W2, W0f, W1f, W2f, ccw);

    const int nblocks = (NPTS + ROWS_PER_BLOCK - 1) / ROWS_PER_BLOCK;
    umnn_mlp<<<nblocks, 128, 0, stream>>>(x, h, b0, b1, b2, W3, b3,
                                          W0f, W1f, W2f, F, outJ);

    umnn_reduce<<<(BD + 255) / 256, 256, 0, stream>>>(x, h, F, ccw, outZ);
}
